// GOLALayer_31997506355866
// MI455X (gfx1250) — matrix operand order repaired and checked by reading, not compiled, not run on hardware
//
#include <hip/hip_runtime.h>

typedef __attribute__((ext_vector_type(16))) _Float16 v16h;
typedef __attribute__((ext_vector_type(8)))  float    v8f;
typedef int v4i __attribute__((vector_size(16)));   // matches builtin's param type

#define HID 128

// ---------------------------------------------------------------------------
// Async global->LDS copy (gfx1250 GLOBAL_LOAD_ASYNC_TO_LDS_B128, ASYNCcnt)
// Probe-derived signature: (v4i AS1* src, v4i AS3* dst, imm offset, imm cpol)
// ---------------------------------------------------------------------------
#if defined(__has_builtin)
#if __has_builtin(__builtin_amdgcn_global_load_async_to_lds_b128)
#define EDGE_ASYNC_COPY 1
#endif
#endif

static __device__ __forceinline__ void async_cp16(const _Float16* g, _Float16* l) {
#if defined(EDGE_ASYNC_COPY)
  __builtin_amdgcn_global_load_async_to_lds_b128(
      (__attribute__((address_space(1))) v4i*)(void*)g,
      (__attribute__((address_space(3))) v4i*)(void*)l, 0, 0);
#else
  *(uint4*)l = *(const uint4*)g;
#endif
}

static __device__ __forceinline__ void wait_async_copies() {
#if defined(EDGE_ASYNC_COPY)
#if __has_builtin(__builtin_amdgcn_s_wait_asynccnt)
  __builtin_amdgcn_s_wait_asynccnt(0);
#else
  asm volatile("s_wait_asynccnt 0" ::: "memory");
#endif
#endif
}

// ---------------------------------------------------------------------------
// WMMA helpers (gfx1250, wave32): D = A(16x32 f16) * B(32x16 f16) + C(16x16 f32)
// ---------------------------------------------------------------------------
static __device__ __forceinline__ v8f wmma_f16(v16h a, v16h b, v8f c) {
  // 8 args: (neg_a, A, neg_b, B, c_mod, C, reuse_a, reuse_b)
  return __builtin_amdgcn_wmma_f32_16x16x32_f16(false, a, false, b, (short)0, c,
                                                false, false);
}

// A fragment from an LDS row-major panel [16][row_stride] halves.
// ISA layout: lane l holds row m=l&15; elems 0..7 = K(kbase + (l>=16?8:0) + 0..7),
// elems 8..15 = K(+16 further).
static __device__ __forceinline__ v16h load_A_frag(const _Float16* p,
                                                   int row_stride, int kbase,
                                                   int lane) {
  int m = lane & 15;
  int kh = (lane >> 4) << 3;
  const _Float16* p0 = p + m * row_stride + kbase + kh;
  v16h a;
#pragma unroll
  for (int i = 0; i < 8; ++i) { a[i] = p0[i]; a[8 + i] = p0[16 + i]; }
  return a;
}

// B fragments pre-shuffled: frag[((kc*8+nt)*32 + lane)*16], contiguous 32B.
static __device__ __forceinline__ v16h load_B_frag(const _Float16* base, int kc,
                                                   int nt, int lane) {
  return *(const v16h*)(base + ((((kc << 3) + nt) << 5) + lane) * 16);
}

static __device__ __forceinline__ float silu(float x) {
  return x / (1.0f + __expf(-x));
}

// Monotonic-int trick float atomic max (lowers to global_atomic_{max_i32,min_u32})
static __device__ __forceinline__ void atomicMaxFloat(float* addr, float val) {
  if (!(val < 0.0f)) atomicMax((int*)addr, __float_as_int(val));
  else               atomicMin((unsigned int*)addr, __float_as_uint(val));
}

// ---------------------------------------------------------------------------
// Prep kernels
// ---------------------------------------------------------------------------
__global__ void convert_h_kernel(const float* __restrict__ h,
                                 _Float16* __restrict__ h16,
                                 float* __restrict__ out, long long n) {
  long long i = (long long)blockIdx.x * 256 + threadIdx.x;
  if (i < n) { float v = h[i]; h16[i] = (_Float16)v; out[i] = v; }
}

__global__ void init_nodes_kernel(float* __restrict__ nmax,
                                  float* __restrict__ nden, long long n) {
  long long i = (long long)blockIdx.x * 256 + threadIdx.x;
  if (i < n) { nmax[i] = -__builtin_inff(); nden[i] = 0.0f; }
}

// Shuffle a row-major [krows x ncols] f32 weight (zero-padded) into per-lane
// B-fragment layout: frag[kc][nt][lane][e] = W[kc*32+koff+(e&7)+(e>=8?16:0)][nt*16+(lane&15)]
__global__ void prep_frag_kernel(const float* __restrict__ W, int krows,
                                 int ncols, int ntcount, int total,
                                 _Float16* __restrict__ frag) {
  int t = blockIdx.x * blockDim.x + threadIdx.x;
  if (t >= total) return;
  int e    = t & 15;
  int lane = (t >> 4) & 31;
  int rest = t >> 9;
  int nt   = rest % ntcount;
  int kc   = rest / ntcount;
  int koff = (lane < 16) ? 0 : 8;
  int kk   = kc * 32 + koff + (e & 7) + ((e >> 3) << 4);
  int n    = nt * 16 + (lane & 15);
  frag[t] = (kk < krows && n < ncols) ? (_Float16)W[kk * ncols + n]
                                      : (_Float16)0.0f;
}

// ---------------------------------------------------------------------------
// hv = h @ Wv  (per-node value projection, WMMA, contiguous 16-node tiles)
// ---------------------------------------------------------------------------
__global__ void __launch_bounds__(128)
hv_wmma_kernel(const _Float16* __restrict__ h16, const _Float16* __restrict__ Wvf,
               float* __restrict__ hv, long long N, long long ntiles) {
  __shared__ __align__(16) _Float16 sm[4][16 * HID];
  int wave = threadIdx.x >> 5, lane = threadIdx.x & 31;
  long long tile = (long long)blockIdx.x * 4 + wave;
  if (tile >= ntiles) return;
  _Float16* A = sm[wave];
  long long base = tile * 16;
  for (int i = lane; i < 256; i += 32) {   // 16 rows x 16 segs of 8 halves (16B)
    int r = i >> 4, seg = i & 15;
    long long node = base + r; if (node >= N) node = N - 1;
    *(uint4*)&A[r * HID + seg * 8] = *(const uint4*)&h16[node * HID + seg * 8];
  }
  v8f acc[8];
#pragma unroll
  for (int nt = 0; nt < 8; ++nt)
#pragma unroll
    for (int i = 0; i < 8; ++i) acc[nt][i] = 0.0f;
#pragma unroll
  for (int kc = 0; kc < 4; ++kc) {
    v16h a = load_A_frag(A, HID, kc * 32, lane);
#pragma unroll
    for (int nt = 0; nt < 8; ++nt)
      acc[nt] = wmma_f16(a, load_B_frag(Wvf, kc, nt, lane), acc[nt]);
  }
  int col = lane & 15, rbase = (lane >> 4) << 3;
#pragma unroll
  for (int nt = 0; nt < 8; ++nt)
#pragma unroll
    for (int r = 0; r < 8; ++r) {
      long long node = base + rbase + r;
      if (node < N) hv[node * HID + nt * 16 + col] = acc[nt][r];
    }
}

// ---------------------------------------------------------------------------
// Per-edge score MLP: 16 edges per wave. Async-DMA gather into LDS, WMMA for
// all three layers (layer 3 uses a zero-padded 128x16 W3 panel, column 0 live).
// Writes scores[] and atomic-max into node_max[dst].
// ---------------------------------------------------------------------------
__global__ void __launch_bounds__(128)
edge_mlp_kernel(const _Float16* __restrict__ h16, const long long* __restrict__ eidx,
                const float* __restrict__ rel, const float* __restrict__ dist,
                const float* __restrict__ b1, const float* __restrict__ b2,
                const float* __restrict__ b3,
                const _Float16* __restrict__ W1f, const _Float16* __restrict__ W2f,
                const _Float16* __restrict__ W3f,
                float* __restrict__ scores, float* __restrict__ node_max,
                long long E, long long ntiles) {
  // per-wave LDS halves: Adst 2048 | Asrc 2048 | Ageo 512 | Xp 2048 | idx 64
  __shared__ __align__(16) _Float16 sm[4][6720];
  int wave = threadIdx.x >> 5, lane = threadIdx.x & 31;
  long long tile = (long long)blockIdx.x * 4 + wave;
  if (tile >= ntiles) return;
  _Float16* Adst = sm[wave];
  _Float16* Asrc = Adst + 2048;
  _Float16* Ageo = Adst + 4096;
  _Float16* Xp   = Adst + 4608;
  int*      idx  = (int*)(Adst + 6656);   // [0..15]=dst, [16..31]=src
  const long long* dstI = eidx;
  const long long* srcI = eidx + E;
  long long e0 = tile * 16;

  // prefetch next tile's index words into GL2 (global_prefetch_b8)
  __builtin_prefetch((const void*)&dstI[e0 + 16], 0, 1);
  __builtin_prefetch((const void*)&srcI[e0 + 16], 0, 1);

  // stage edge indices once (int32) in LDS
  {
    long long e = e0 + (lane & 15); if (e >= E) e = E - 1;
    idx[lane] = (int)((lane < 16) ? dstI[e] : srcI[e]);
  }

  // gather h[dst], h[src] rows straight into LDS (async, 16B segments)
  for (int i = lane; i < 256; i += 32) {
    int r = i >> 4, seg = i & 15;
    async_cp16(&h16[(long long)idx[r] * HID + seg * 8],
               &Adst[r * HID + seg * 8]);
    async_cp16(&h16[(long long)idx[16 + r] * HID + seg * 8],
               &Asrc[r * HID + seg * 8]);
  }
  // geo panel: K=0..2 rel_pos, K=3 distance, K=4..31 zero (pads 260 -> 288)
  for (int i = lane; i < 512; i += 32) {
    int r = i >> 5, k = i & 31;
    long long e = e0 + r; if (e >= E) e = E - 1;
    float v = 0.0f;
    if (k < 3) v = rel[e * 3 + k];
    else if (k == 3) v = dist[e];
    Ageo[r * 32 + k] = (_Float16)v;
  }
  wait_async_copies();

  int col = lane & 15, rbase = (lane >> 4) << 3;

  // ---- layer 1: acc = bias; += [h_dst | h_src | geo] @ W1  (72 WMMAs) ----
  v8f acc[8];
#pragma unroll
  for (int nt = 0; nt < 8; ++nt) {
    float bv = b1[nt * 16 + col];
#pragma unroll
    for (int i = 0; i < 8; ++i) acc[nt][i] = bv;
  }
#pragma unroll
  for (int kc = 0; kc < 4; ++kc) {
    v16h a = load_A_frag(Adst, HID, kc * 32, lane);
#pragma unroll
    for (int nt = 0; nt < 8; ++nt)
      acc[nt] = wmma_f16(a, load_B_frag(W1f, kc, nt, lane), acc[nt]);
  }
#pragma unroll
  for (int kc = 0; kc < 4; ++kc) {
    v16h a = load_A_frag(Asrc, HID, kc * 32, lane);
#pragma unroll
    for (int nt = 0; nt < 8; ++nt)
      acc[nt] = wmma_f16(a, load_B_frag(W1f, 4 + kc, nt, lane), acc[nt]);
  }
  {
    v16h a = load_A_frag(Ageo, 32, 0, lane);
#pragma unroll
    for (int nt = 0; nt < 8; ++nt)
      acc[nt] = wmma_f16(a, load_B_frag(W1f, 8, nt, lane), acc[nt]);
  }
  // silu -> Xp (f16 A-panel for layer 2)
#pragma unroll
  for (int nt = 0; nt < 8; ++nt)
#pragma unroll
    for (int r = 0; r < 8; ++r)
      Xp[(rbase + r) * HID + nt * 16 + col] = (_Float16)silu(acc[nt][r]);

  // ---- layer 2: 32 WMMAs ----
#pragma unroll
  for (int nt = 0; nt < 8; ++nt) {
    float bv = b2[nt * 16 + col];
#pragma unroll
    for (int i = 0; i < 8; ++i) acc[nt][i] = bv;
  }
#pragma unroll
  for (int kc = 0; kc < 4; ++kc) {
    v16h a = load_A_frag(Xp, HID, kc * 32, lane);
#pragma unroll
    for (int nt = 0; nt < 8; ++nt)
      acc[nt] = wmma_f16(a, load_B_frag(W2f, kc, nt, lane), acc[nt]);
  }
  _Float16* X2 = Adst;  // reuse gather panel
#pragma unroll
  for (int nt = 0; nt < 8; ++nt)
#pragma unroll
    for (int r = 0; r < 8; ++r)
      X2[(rbase + r) * HID + nt * 16 + col] = (_Float16)silu(acc[nt][r]);

  // ---- layer 3 via WMMA: scores = X2 @ W3panel (4 WMMAs, col 0 live) ----
  v8f sacc;
#pragma unroll
  for (int i = 0; i < 8; ++i) sacc[i] = b3[0];
#pragma unroll
  for (int kc = 0; kc < 4; ++kc) {
    v16h a = load_A_frag(X2, HID, kc * 32, lane);
    v16h b = *(const v16h*)(W3f + ((kc << 5) + lane) * 16);
    sacc = wmma_f16(a, b, sacc);
  }
  if ((lane & 15) == 0) {   // lanes 0 and 16 hold column 0 (rows 0..7 / 8..15)
#pragma unroll
    for (int r = 0; r < 8; ++r) {
      long long e = e0 + rbase + r;
      if (e < E) {
        float s = sacc[r];
        scores[e] = s;
        atomicMaxFloat(&node_max[idx[rbase + r]], s);
      }
    }
  }
}

// ---------------------------------------------------------------------------
// exp(score - max[dst]) and denominator accumulation
// ---------------------------------------------------------------------------
__global__ void exp_den_kernel(const float* __restrict__ scores,
                               const long long* __restrict__ eidx,
                               const float* __restrict__ nmax,
                               float* __restrict__ nden,
                               float* __restrict__ exps, long long E) {
  long long i = (long long)blockIdx.x * 256 + threadIdx.x;
  if (i >= E) return;
  long long d = eidx[i];
  float ex = __expf(scores[i] - nmax[d]);
  exps[i] = ex;
  atomicAdd(&nden[d], ex);
}

// ---------------------------------------------------------------------------
// out[dst] += hv[src] * (alpha * node_weight[src]);  32 lanes per edge
// ---------------------------------------------------------------------------
__global__ void scatter_kernel(const long long* __restrict__ eidx,
                               const float* __restrict__ exps,
                               const float* __restrict__ nden,
                               const float* __restrict__ nodew,
                               const float* __restrict__ hv,
                               float* __restrict__ out, long long E) {
  long long e = (long long)blockIdx.x * 8 + (threadIdx.x >> 5);
  int lane = threadIdx.x & 31;
  if (e >= E) return;
  long long d = eidx[e];
  long long s = eidx[E + e];
  float w = exps[e] / (nden[d] + 1e-12f) * nodew[s];
  const float4 v = *(const float4*)&hv[s * HID + lane * 4];
  float* o = out + d * HID + lane * 4;
  atomicAdd(o + 0, v.x * w);
  atomicAdd(o + 1, v.y * w);
  atomicAdd(o + 2, v.z * w);
  atomicAdd(o + 3, v.w * w);
}

// ---------------------------------------------------------------------------
static inline size_t align256(size_t x) { return (x + 255) & ~(size_t)255; }

extern "C" void kernel_launch(void* const* d_in, const int* in_sizes, int n_in,
                              void* d_out, int out_size, void* d_ws, size_t ws_size,
                              hipStream_t stream) {
  const float*     h     = (const float*)d_in[0];
  const long long* eidx  = (const long long*)d_in[1];  // int64 per reference
  const float*     rel   = (const float*)d_in[2];
  const float*     dist  = (const float*)d_in[3];
  const float*     nodew = (const float*)d_in[4];
  const float*     W1    = (const float*)d_in[5];
  const float*     b1    = (const float*)d_in[6];
  const float*     W2    = (const float*)d_in[7];
  const float*     b2    = (const float*)d_in[8];
  const float*     W3    = (const float*)d_in[9];
  const float*     b3    = (const float*)d_in[10];
  const float*     Wv    = (const float*)d_in[11];
  float* out = (float*)d_out;

  long long N = (long long)in_sizes[0] / HID;
  long long E = (long long)in_sizes[1] / 2;

  // workspace carve-out
  char* ws = (char*)d_ws;
  size_t off = 0;
  _Float16* h16 = (_Float16*)(ws + off); off = align256(off + (size_t)N * HID * 2);
  float* hv     = (float*)(ws + off);    off = align256(off + (size_t)N * HID * 4);
  float* scores = (float*)(ws + off);    off = align256(off + (size_t)E * 4);
  float* exps   = (float*)(ws + off);    off = align256(off + (size_t)E * 4);
  float* nmax   = (float*)(ws + off);    off = align256(off + (size_t)N * 4);
  float* nden   = (float*)(ws + off);    off = align256(off + (size_t)N * 4);
  _Float16* W1f = (_Float16*)(ws + off); off = align256(off + (size_t)36864 * 2);
  _Float16* W2f = (_Float16*)(ws + off); off = align256(off + (size_t)16384 * 2);
  _Float16* Wvf = (_Float16*)(ws + off); off = align256(off + (size_t)16384 * 2);
  _Float16* W3f = (_Float16*)(ws + off); off = align256(off + (size_t)2048 * 2);
  (void)ws_size; (void)n_in; (void)out_size;

  long long NH = N * HID;
  convert_h_kernel<<<(unsigned)((NH + 255) / 256), 256, 0, stream>>>(h, h16, out, NH);
  init_nodes_kernel<<<(unsigned)((N + 255) / 256), 256, 0, stream>>>(nmax, nden, N);
  prep_frag_kernel<<<(36864 + 255) / 256, 256, 0, stream>>>(W1, 260, 128, 8, 36864, W1f);
  prep_frag_kernel<<<(16384 + 255) / 256, 256, 0, stream>>>(W2, 128, 128, 8, 16384, W2f);
  prep_frag_kernel<<<(16384 + 255) / 256, 256, 0, stream>>>(Wv, 128, 128, 8, 16384, Wvf);
  prep_frag_kernel<<<(2048 + 255) / 256, 256, 0, stream>>>(W3, 128, 1, 1, 2048, W3f);

  long long ntN = (N + 15) / 16;
  hv_wmma_kernel<<<(unsigned)((ntN + 3) / 4), 128, 0, stream>>>(h16, Wvf, hv, N, ntN);

  long long ntE = (E + 15) / 16;
  edge_mlp_kernel<<<(unsigned)((ntE + 3) / 4), 128, 0, stream>>>(
      h16, eidx, rel, dist, b1, b2, b3, W1f, W2f, W3f, scores, nmax, E, ntE);

  exp_den_kernel<<<(unsigned)((E + 255) / 256), 256, 0, stream>>>(scores, eidx, nmax,
                                                                  nden, exps, E);
  scatter_kernel<<<(unsigned)((E + 7) / 8), 256, 0, stream>>>(eidx, exps, nden, nodew,
                                                              hv, out, E);
}
